// CrossAttention_47450798686575
// MI455X (gfx1250) — compile-verified, matrix operand order checked by reading, not run on hardware
//
#include <hip/hip_runtime.h>
#include <hip/hip_bf16.h>

typedef _Float16 half_t;
typedef __attribute__((ext_vector_type(16))) _Float16 v16h;
typedef __attribute__((ext_vector_type(8)))  _Float16 v8h;
typedef __attribute__((ext_vector_type(8)))  float    v8f;
typedef __attribute__((ext_vector_type(4)))  float    v4f;

#define B_      16
#define N_      256
#define M_      1024
#define HEADS_  8
#define DHEAD_  96
#define INNER_  768
#define DDIM_   768
#define TDIM_   2560
#define NEGVAL  (-1000000.0f)
#define LN_EPS  1e-5f

// wave32 16-bit fragment: vector elements 0..7 hold K = kb+8*hi .. +7,
// elements 8..15 hold K = kb+16+8*hi .. +7  (contiguous 8-element runs).

// ---------------------------------------------------------------------------
// Weight pre-transpose: Wt[n*K + k] = (f16) W[k*N + n]
// ---------------------------------------------------------------------------
__global__ __launch_bounds__(256)
void transpose_w(const float* __restrict__ W, half_t* __restrict__ Wt,
                 int K, int N)
{
    const int k = blockIdx.x * 32 + (threadIdx.x & 31);
    const int n = blockIdx.y * 8 + (threadIdx.x >> 5);
    if (k < K && n < N)
        Wt[(size_t)n * K + k] = (half_t)W[(size_t)k * N + n];
}

// Build a v16h fragment from fp32 source (row-major in K), base = row start.
__device__ __forceinline__ v16h frag_from_f32(const float* __restrict__ p,
                                              int kb, int hi)
{
    const v4f x0 = *(const v4f*)(p + kb + 8 * hi);
    const v4f x1 = *(const v4f*)(p + kb + 8 * hi + 4);
    const v4f x2 = *(const v4f*)(p + kb + 16 + 8 * hi);
    const v4f x3 = *(const v4f*)(p + kb + 16 + 8 * hi + 4);
    v16h a;
#pragma unroll
    for (int e = 0; e < 4; ++e) {
        a[e]      = (half_t)x0[e];
        a[e + 4]  = (half_t)x1[e];
        a[e + 8]  = (half_t)x2[e];
        a[e + 12] = (half_t)x3[e];
    }
    return a;
}

// Build a v16h fragment from f16 source (contiguous in K), base = row start.
__device__ __forceinline__ v16h frag_from_h(const half_t* __restrict__ p,
                                            int kb, int hi)
{
    const v8h w0 = *(const v8h*)(p + kb + 8 * hi);
    const v8h w1 = *(const v8h*)(p + kb + 16 + 8 * hi);
    v16h a;
#pragma unroll
    for (int e = 0; e < 8; ++e) { a[e] = w0[e]; a[e + 8] = w1[e]; }
    return a;
}

// ---------------------------------------------------------------------------
// GEMM  Y(f16, row-major) = X(fp32, RxK) @ Wt^T (Wt stored as [n][k], f16)
// One wave computes a 16x64 tile: 4 accumulators share each A fragment.
// Block = 128 threads = 4 waves = 4 row-tiles.
// ---------------------------------------------------------------------------
__global__ __launch_bounds__(128)
void proj_gemm(const float* __restrict__ X,
               const half_t* __restrict__ Wt,   // N x K, K contiguous
               half_t* __restrict__ Y,          // R x N
               int K, int N)
{
    const int wid  = threadIdx.x >> 5;
    const int lane = threadIdx.x & 31;
    const int lo = lane & 15, hi = lane >> 4;
    const int mrow = (blockIdx.x * 4 + wid) * 16;
    const int ncol = blockIdx.y * 64;

    const float* xrow = X + (size_t)(mrow + lo) * K;
    const half_t* wrow0 = Wt + (size_t)(ncol + 0  + lo) * K;
    const half_t* wrow1 = Wt + (size_t)(ncol + 16 + lo) * K;
    const half_t* wrow2 = Wt + (size_t)(ncol + 32 + lo) * K;
    const half_t* wrow3 = Wt + (size_t)(ncol + 48 + lo) * K;

    v8f c0 = {}, c1 = {}, c2 = {}, c3 = {};
    for (int kb = 0; kb < K; kb += 32) {
        const v16h a  = frag_from_f32(xrow, kb, hi);
        const v16h b0 = frag_from_h(wrow0, kb, hi);
        const v16h b1 = frag_from_h(wrow1, kb, hi);
        const v16h b2 = frag_from_h(wrow2, kb, hi);
        const v16h b3 = frag_from_h(wrow3, kb, hi);
        c0 = __builtin_amdgcn_wmma_f32_16x16x32_f16(false, a, false, b0,
                                                    (short)0, c0, false, false);
        c1 = __builtin_amdgcn_wmma_f32_16x16x32_f16(false, a, false, b1,
                                                    (short)0, c1, false, false);
        c2 = __builtin_amdgcn_wmma_f32_16x16x32_f16(false, a, false, b2,
                                                    (short)0, c2, false, false);
        c3 = __builtin_amdgcn_wmma_f32_16x16x32_f16(false, a, false, b3,
                                                    (short)0, c3, false, false);
        if (kb + 32 < K) {
            __builtin_prefetch(xrow + kb + 32, 0, 1);
            __builtin_prefetch(wrow0 + kb + 32, 0, 1);
        }
    }
#pragma unroll
    for (int r = 0; r < 8; ++r) {
        const size_t rbase = (size_t)(mrow + r + 8 * hi) * N + ncol + lo;
        Y[rbase + 0]  = (half_t)c0[r];
        Y[rbase + 16] = (half_t)c1[r];
        Y[rbase + 32] = (half_t)c2[r];
        Y[rbase + 48] = (half_t)c3[r];
    }
}

// Same GEMM but stores result transposed per (b,h):  Vt[(b,h,d)][m]
__global__ __launch_bounds__(128)
void proj_gemm_vT(const float* __restrict__ X,   // (B*M) x K
                  const half_t* __restrict__ Wt, // INNER x K
                  half_t* __restrict__ Yt,       // (B*H*DHEAD) x M
                  int K)
{
    const int wid  = threadIdx.x >> 5;
    const int lane = threadIdx.x & 31;
    const int lo = lane & 15, hi = lane >> 4;
    const int mrow = (blockIdx.x * 4 + wid) * 16;  // over B*M
    const int ncol = blockIdx.y * 64;              // over INNER

    const float* xrow = X + (size_t)(mrow + lo) * K;
    const half_t* wrow0 = Wt + (size_t)(ncol + 0  + lo) * K;
    const half_t* wrow1 = Wt + (size_t)(ncol + 16 + lo) * K;
    const half_t* wrow2 = Wt + (size_t)(ncol + 32 + lo) * K;
    const half_t* wrow3 = Wt + (size_t)(ncol + 48 + lo) * K;

    v8f c0 = {}, c1 = {}, c2 = {}, c3 = {};
    for (int kb = 0; kb < K; kb += 32) {
        const v16h a  = frag_from_f32(xrow, kb, hi);
        const v16h b0 = frag_from_h(wrow0, kb, hi);
        const v16h b1 = frag_from_h(wrow1, kb, hi);
        const v16h b2 = frag_from_h(wrow2, kb, hi);
        const v16h b3 = frag_from_h(wrow3, kb, hi);
        c0 = __builtin_amdgcn_wmma_f32_16x16x32_f16(false, a, false, b0,
                                                    (short)0, c0, false, false);
        c1 = __builtin_amdgcn_wmma_f32_16x16x32_f16(false, a, false, b1,
                                                    (short)0, c1, false, false);
        c2 = __builtin_amdgcn_wmma_f32_16x16x32_f16(false, a, false, b2,
                                                    (short)0, c2, false, false);
        c3 = __builtin_amdgcn_wmma_f32_16x16x32_f16(false, a, false, b3,
                                                    (short)0, c3, false, false);
        if (kb + 32 < K) {
            __builtin_prefetch(xrow + kb + 32, 0, 1);
            __builtin_prefetch(wrow0 + kb + 32, 0, 1);
        }
    }
#pragma unroll
    for (int j = 0; j < 4; ++j) {
        const v8f cj = (j == 0) ? c0 : (j == 1) ? c1 : (j == 2) ? c2 : c3;
        const int col = ncol + j * 16 + lo;        // within INNER
        const int hh = col / DHEAD_, d = col % DHEAD_;
#pragma unroll
        for (int r = 0; r < 8; ++r) {
            const int row = mrow + r + 8 * hi;     // over B*M
            const int b = row >> 10, m = row & (M_ - 1);
            Yt[(((size_t)b * HEADS_ + hh) * DHEAD_ + d) * M_ + m] = (half_t)cj[r];
        }
    }
}

// ---------------------------------------------------------------------------
// Attention: one workgroup = (batch b, head h, 16-row q tile).
// ---------------------------------------------------------------------------
__global__ __launch_bounds__(256)
void attn_kernel(const half_t* __restrict__ Qh,   // (B*N) x INNER
                 const half_t* __restrict__ Kh,   // (B*M) x INNER
                 const half_t* __restrict__ Vt,   // (B*H*DHEAD) x M
                 const int* __restrict__ dmask,   // (B, N)
                 const int* __restrict__ pmask,   // (B, M)
                 float* __restrict__ attn,        // (B, H, N, M) fp32 output
                 half_t* __restrict__ ctx)        // (B*N) x INNER
{
    __shared__ float sdots[16 * M_];              // 64 KB

    const int wid  = threadIdx.x >> 5;
    const int lane = threadIdx.x & 31;
    const int lo = lane & 15, hi = lane >> 4;

    const int ntile = blockIdx.x & 15;
    const int h     = (blockIdx.x >> 4) & 7;
    const int b     = blockIdx.x >> 7;

    const int n0 = ntile * 16;
    const float scale = 0.10206207261596577f;     // 96^-0.5

    // Q fragments: 3 chunks of 32 cover d=96 (contiguous b128 loads)
    const half_t* qrow = Qh + ((size_t)b * N_ + n0 + lo) * INNER_
                            + (size_t)h * DHEAD_;
    const v16h qa0 = frag_from_h(qrow, 0,  hi);
    const v16h qa1 = frag_from_h(qrow, 32, hi);
    const v16h qa2 = frag_from_h(qrow, 64, hi);

    // -------- scores: dots[16 x 1024] = (Q K^T) * scale, masked ------------
    for (int mt = wid; mt < M_ / 16; mt += 8) {
        const int m0 = mt * 16;
        const half_t* krow = Kh + ((size_t)b * M_ + m0 + lo) * INNER_
                                + (size_t)h * DHEAD_;
        v8f c = {};
        c = __builtin_amdgcn_wmma_f32_16x16x32_f16(false, qa0, false,
              frag_from_h(krow, 0, hi),  (short)0, c, false, false);
        c = __builtin_amdgcn_wmma_f32_16x16x32_f16(false, qa1, false,
              frag_from_h(krow, 32, hi), (short)0, c, false, false);
        c = __builtin_amdgcn_wmma_f32_16x16x32_f16(false, qa2, false,
              frag_from_h(krow, 64, hi), (short)0, c, false, false);

        const int colv = pmask[b * M_ + m0 + lo];
#pragma unroll
        for (int r = 0; r < 8; ++r) {
            const int rr = r + 8 * hi;
            const int rowv = dmask[b * N_ + n0 + rr];
            sdots[rr * M_ + m0 + lo] =
                (rowv != 0 && colv != 0) ? c[r] * scale : NEGVAL;
        }
    }
    __syncthreads();

    // -------- softmax per row (wave32 shuffle reductions) ------------------
    for (int row = wid; row < 16; row += 8) {
        float lmax = -3.0e38f;
        for (int m = lane; m < M_; m += 32)
            lmax = fmaxf(lmax, sdots[row * M_ + m]);
#pragma unroll
        for (int s = 16; s; s >>= 1) lmax = fmaxf(lmax, __shfl_xor(lmax, s, 32));

        float lsum = 0.f;
        for (int m = lane; m < M_; m += 32) {
            const float e = __expf(sdots[row * M_ + m] - lmax);
            sdots[row * M_ + m] = e;
            lsum += e;
        }
#pragma unroll
        for (int s = 16; s; s >>= 1) lsum += __shfl_xor(lsum, s, 32);

        const float inv = 1.0f / lsum;
        const size_t abase = (((size_t)b * HEADS_ + h) * N_ + n0 + row) * M_;
        for (int m = lane; m < M_; m += 32) {
            const float p = sdots[row * M_ + m] * inv;
            attn[abase + m]     = p;   // fp32 attn output
            sdots[row * M_ + m] = p;   // probs stay in LDS for P@V
        }
    }
    __syncthreads();

    // -------- context = P @ V : 6 d-tiles, accumulate over m=1024 ----------
    if (wid < 6) {
        const int d0 = wid * 16;
        const float*  prow = sdots + lo * M_;                  // probs row (LDS)
        const half_t* vrow = Vt + (((size_t)b * HEADS_ + h) * DHEAD_
                                    + d0 + lo) * M_;           // V col, m-contig
        v8f c = {};
        for (int kb = 0; kb < M_; kb += 32) {
            // A: probs (fp32 in LDS, contiguous in m) -> ds_load_b128 x4
            const v4f p0 = *(const v4f*)(prow + kb + 8 * hi);
            const v4f p1 = *(const v4f*)(prow + kb + 8 * hi + 4);
            const v4f p2 = *(const v4f*)(prow + kb + 16 + 8 * hi);
            const v4f p3 = *(const v4f*)(prow + kb + 16 + 8 * hi + 4);
            v16h a;
#pragma unroll
            for (int e = 0; e < 4; ++e) {
                a[e]      = (half_t)p0[e];
                a[e + 4]  = (half_t)p1[e];
                a[e + 8]  = (half_t)p2[e];
                a[e + 12] = (half_t)p3[e];
            }
            const v16h bf = frag_from_h(vrow, kb, hi);
            c = __builtin_amdgcn_wmma_f32_16x16x32_f16(false, a, false, bf,
                                                       (short)0, c, false, false);
        }
#pragma unroll
        for (int r = 0; r < 8; ++r) {
            const int rr = r + 8 * hi;
            ctx[((size_t)b * N_ + n0 + rr) * INNER_ +
                (size_t)h * DHEAD_ + d0 + lo] = (half_t)c[r];
        }
    }
}

// ---------------------------------------------------------------------------
// Out-projection + bias + residual + LayerNorm. One workgroup per 16 rows.
// Each wave computes 16x64 via 4 shared-A accumulators.
// ---------------------------------------------------------------------------
__global__ __launch_bounds__(256)
void outproj_ln_kernel(const half_t* __restrict__ ctx,  // (B*N) x INNER
                       const half_t* __restrict__ WoT,  // DDIM x INNER (k-contig)
                       const float* __restrict__ bo,
                       const float* __restrict__ drug,  // (B*N) x DDIM
                       const float* __restrict__ gamma,
                       const float* __restrict__ beta,
                       float* __restrict__ out)         // (B*N) x DDIM
{
    __shared__ float sx[16 * DDIM_];                    // 48 KB

    const int wid  = threadIdx.x >> 5;
    const int lane = threadIdx.x & 31;
    const int lo = lane & 15, hi = lane >> 4;
    const int row0 = blockIdx.x * 16;

    const half_t* arow = ctx + (size_t)(row0 + lo) * INNER_;

    for (int g = wid; g < DDIM_ / 64; g += 8) {         // 12 groups of 64 cols
        const int ncol = g * 64;
        const half_t* wrow0 = WoT + (size_t)(ncol + 0  + lo) * INNER_;
        const half_t* wrow1 = WoT + (size_t)(ncol + 16 + lo) * INNER_;
        const half_t* wrow2 = WoT + (size_t)(ncol + 32 + lo) * INNER_;
        const half_t* wrow3 = WoT + (size_t)(ncol + 48 + lo) * INNER_;
        v8f c0 = {}, c1 = {}, c2 = {}, c3 = {};
        for (int kb = 0; kb < INNER_; kb += 32) {
            const v16h a  = frag_from_h(arow, kb, hi);
            c0 = __builtin_amdgcn_wmma_f32_16x16x32_f16(false, a, false,
                   frag_from_h(wrow0, kb, hi), (short)0, c0, false, false);
            c1 = __builtin_amdgcn_wmma_f32_16x16x32_f16(false, a, false,
                   frag_from_h(wrow1, kb, hi), (short)0, c1, false, false);
            c2 = __builtin_amdgcn_wmma_f32_16x16x32_f16(false, a, false,
                   frag_from_h(wrow2, kb, hi), (short)0, c2, false, false);
            c3 = __builtin_amdgcn_wmma_f32_16x16x32_f16(false, a, false,
                   frag_from_h(wrow3, kb, hi), (short)0, c3, false, false);
        }
#pragma unroll
        for (int j = 0; j < 4; ++j) {
            const v8f cj = (j == 0) ? c0 : (j == 1) ? c1 : (j == 2) ? c2 : c3;
            const int cc = ncol + j * 16 + lo;
#pragma unroll
            for (int r = 0; r < 8; ++r) {
                const int rr = r + 8 * hi;
                sx[rr * DDIM_ + cc] =
                    cj[r] + bo[cc] +
                    drug[(size_t)(row0 + rr) * DDIM_ + cc];
            }
        }
    }
    __syncthreads();

    for (int row = wid; row < 16; row += 8) {
        float s = 0.f;
        for (int cc = lane; cc < DDIM_; cc += 32) s += sx[row * DDIM_ + cc];
#pragma unroll
        for (int d = 16; d; d >>= 1) s += __shfl_xor(s, d, 32);
        const float mu = s * (1.0f / DDIM_);

        float vs = 0.f;
        for (int cc = lane; cc < DDIM_; cc += 32) {
            const float dv = sx[row * DDIM_ + cc] - mu;
            vs += dv * dv;
        }
#pragma unroll
        for (int d = 16; d; d >>= 1) vs += __shfl_xor(vs, d, 32);
        const float rstd = rsqrtf(vs * (1.0f / DDIM_) + LN_EPS);

        for (int cc = lane; cc < DDIM_; cc += 32)
            out[(size_t)(row0 + row) * DDIM_ + cc] =
                (sx[row * DDIM_ + cc] - mu) * rstd * gamma[cc] + beta[cc];
    }
}

// ---------------------------------------------------------------------------
extern "C" void kernel_launch(void* const* d_in, const int* in_sizes, int n_in,
                              void* d_out, int out_size, void* d_ws, size_t ws_size,
                              hipStream_t stream) {
    (void)in_sizes; (void)n_in; (void)out_size; (void)ws_size;

    const float* drug   = (const float*)d_in[0];
    const float* target = (const float*)d_in[1];
    const int*   dmask  = (const int*)d_in[2];
    const int*   pmask  = (const int*)d_in[3];
    const float* Wq     = (const float*)d_in[4];
    const float* Wk     = (const float*)d_in[5];
    const float* Wv     = (const float*)d_in[6];
    const float* Wo     = (const float*)d_in[7];
    const float* bo     = (const float*)d_in[8];
    const float* gamma  = (const float*)d_in[9];
    const float* beta   = (const float*)d_in[10];

    float* out  = (float*)d_out;                               // (16,256,768)
    float* attn = out + (size_t)B_ * N_ * DDIM_;               // (16,8,256,1024)

    // workspace carve (f16)
    half_t* Qh  = (half_t*)d_ws;                               // (B*N) x INNER
    half_t* Kh  = Qh  + (size_t)B_ * N_ * INNER_;              // (B*M) x INNER
    half_t* Vth = Kh  + (size_t)B_ * M_ * INNER_;              // (B*H*96) x M
    half_t* Ch  = Vth + (size_t)B_ * M_ * INNER_;              // (B*N) x INNER
    half_t* WqT = Ch  + (size_t)B_ * N_ * INNER_;              // INNER x DDIM
    half_t* WkT = WqT + (size_t)INNER_ * DDIM_;                // INNER x TDIM
    half_t* WvT = WkT + (size_t)INNER_ * TDIM_;
    half_t* WoT = WvT + (size_t)INNER_ * TDIM_;                // DDIM x INNER

    // weight transposes (f32 -> f16, K-contiguous per output column)
    transpose_w<<<dim3(DDIM_ / 32, INNER_ / 8), 256, 0, stream>>>(Wq, WqT, DDIM_, INNER_);
    transpose_w<<<dim3(TDIM_ / 32, INNER_ / 8), 256, 0, stream>>>(Wk, WkT, TDIM_, INNER_);
    transpose_w<<<dim3(TDIM_ / 32, INNER_ / 8), 256, 0, stream>>>(Wv, WvT, TDIM_, INNER_);
    transpose_w<<<dim3(INNER_ / 32, DDIM_ / 8), 256, 0, stream>>>(Wo, WoT, INNER_, DDIM_);

    // projections (block = 4 waves = 4 row tiles; each wave does a 16x64 tile)
    proj_gemm<<<dim3((B_ * N_) / 64, INNER_ / 64), 128, 0, stream>>>(
        drug, WqT, Qh, DDIM_, INNER_);
    proj_gemm<<<dim3((B_ * M_) / 64, INNER_ / 64), 128, 0, stream>>>(
        target, WkT, Kh, TDIM_, INNER_);
    proj_gemm_vT<<<dim3((B_ * M_) / 64, INNER_ / 64), 128, 0, stream>>>(
        target, WvT, Vth, TDIM_);

    // attention (scores + softmax + P@V)
    attn_kernel<<<B_ * HEADS_ * (N_ / 16), 256, 0, stream>>>(
        Qh, Kh, Vth, dmask, pmask, attn, Ch);

    // out-projection + residual + LayerNorm
    outproj_ln_kernel<<<(B_ * N_) / 16, 256, 0, stream>>>(
        Ch, WoT, bo, drug, gamma, beta, out);
}
